// SimpleCnnNcpActor_38079180046406
// MI455X (gfx1250) — matrix operand order repaired and checked by reading, not compiled, not run on hardware
//
#include <hip/hip_runtime.h>
#include <hip/hip_bf16.h>
#include <stdint.h>

// ---------------------------------------------------------------------------
// SimpleCnnNcpActor on MI455X (gfx1250, wave32, WMMA)
//   Activations stored f16 NHWC with a 1-pixel zero halo so the WMMA A-operand
//   is built from two contiguous b128 loads per lane per K-step (no predicated
//   scalar gathers, no boundary branches in the GEMM inner loop).
//   conv1 (1->32):  direct VALU conv, NHWC-halo f16 out (4x b128 stores/pixel)
//   conv2 (32->64): implicit-GEMM WMMA f16, M=327680,N=64,K=288 (9 K-steps)
//                   D-tile transposed through LDS -> one b128 store per lane
//   conv3 (64->32): implicit-GEMM WMMA f16, M=81920, N=32,K=576 (18 K-steps)
//   LTC cell: 64 blocks, 8 waves each; lane-strided sensory reduction.
// ---------------------------------------------------------------------------

typedef __attribute__((ext_vector_type(16))) _Float16 v16h;
typedef __attribute__((ext_vector_type(8)))  _Float16 v8h;
typedef __attribute__((ext_vector_type(8)))  float    v8f;

#define NB      64
#define NS      32
#define NIMG    2048
#define SENSN   1283
#define UNITSN  48
#define MOTORN  4
#define NACT    2
#define UNFOLDS 6

// conv geometry (H1P/W1P etc. are haloed NHWC dims)
#define H0 40
#define W0 64
#define H1 20
#define W1 32
#define H1P 22
#define W1P 34
#define H2 10
#define W2 16
#define H2P 12
#define W2P 18
#define H3 5
#define W3 8

__device__ __forceinline__ float sigmoidf_(float x) { return 1.0f / (1.0f + __expf(-x)); }
__device__ __forceinline__ float softplusf_(float x) { return (x > 20.0f) ? x : log1pf(__expf(x)); }

// A-operand K mapping within a K=32 step (ISA 7.12.2, 16-bit A 16x32):
// lane-half h: element j -> K = j + 8h (j<8), K = 16 + (j-8) + 8h (j>=8).
// => two contiguous 8-element runs; with NHWC channels these are two b128 loads.
// B-operand: same K mapping as A (element j -> K = j + 8h (j<8),
// K = 16 + (j-8) + 8h (j>=8)).
__device__ __forceinline__ int kmapB(int j, int h) { return (j & 7) + (h << 3) + ((j >> 3) << 4); }

// ---------------------------------------------------------------------------
// zero-fill (halo regions must be 0; d_ws is poisoned)
// ---------------------------------------------------------------------------
__global__ void fill_zero_kernel(uint4* __restrict__ p) {
    size_t i = (size_t)blockIdx.x * blockDim.x + threadIdx.x;
    uint4 z; z.x = 0; z.y = 0; z.z = 0; z.w = 0;
    p[i] = z;
}

// ---------------------------------------------------------------------------
// conv1: one thread per output pixel, all 32 channels, NHWC-halo f16 out
// ---------------------------------------------------------------------------
__global__ void conv1_kernel(const float* __restrict__ x, const float* __restrict__ w1,
                             const float* __restrict__ b1, _Float16* __restrict__ x1) {
    __shared__ float ws[288];
    __shared__ float bs[32];
    int t = threadIdx.x;
    for (int i = t; i < 288; i += blockDim.x) ws[i] = w1[i];
    if (t < 32) bs[t] = b1[t];
    __syncthreads();
    int gid = blockIdx.x * blockDim.x + t;        // 2048*20*32 threads exactly
    int img = gid / (H1 * W1);
    int rem = gid % (H1 * W1);
    int oy = rem / W1, ox = rem % W1;
    const float* xi = x + (size_t)img * (H0 * W0);
    float in[9];
#pragma unroll
    for (int ky = 0; ky < 3; ++ky)
#pragma unroll
        for (int kx = 0; kx < 3; ++kx) {
            int iy = oy * 2 + ky - 1, ix = ox * 2 + kx - 1;
            in[ky * 3 + kx] = (iy >= 0 && iy < H0 && ix >= 0 && ix < W0) ? xi[iy * W0 + ix] : 0.0f;
        }
    _Float16* dst = x1 + (((size_t)img * H1P + (oy + 1)) * W1P + (ox + 1)) * 32;
#pragma unroll
    for (int c8 = 0; c8 < 4; ++c8) {
        v8h v;
#pragma unroll
        for (int k = 0; k < 8; ++k) {
            int co = c8 * 8 + k;
            float acc = bs[co];
#pragma unroll
            for (int q = 0; q < 9; ++q) acc += ws[co * 9 + q] * in[q];
            v[k] = (_Float16)fmaxf(acc, 0.0f);
        }
        ((v8h*)dst)[c8] = v;
    }
}

// ---------------------------------------------------------------------------
// pack conv2 weights into WMMA B fragments: [ntile(4)][kstep(9)][lane(32)][16]
// K ordering: k = (ky*3+kx)*32 + ci  (one filter tap per K=32 step)
// ---------------------------------------------------------------------------
__global__ void pack_b2_kernel(const float* __restrict__ w2, _Float16* __restrict__ Bp2) {
    int t = blockIdx.x * blockDim.x + threadIdx.x;
    if (t >= 4 * 9 * 32) return;
    int lane = t & 31, ks = (t >> 5) % 9, ntile = t / (9 * 32);
    int h = lane >> 4, n = ntile * 16 + (lane & 15);
    int ky = ks / 3, kx = ks % 3;
    _Float16* dst = Bp2 + (size_t)t * 16;
#pragma unroll
    for (int j = 0; j < 16; ++j) {
        int ci = kmapB(j, h);
        dst[j] = (_Float16)w2[((n * 32 + ci) * 3 + ky) * 3 + kx];
    }
}

// pack conv3 weights: [ntile(2)][kstep(18)][lane(32)][16]; k = tap*64 + ci
__global__ void pack_b3_kernel(const float* __restrict__ w3, _Float16* __restrict__ Bp3) {
    int t = blockIdx.x * blockDim.x + threadIdx.x;
    if (t >= 2 * 18 * 32) return;
    int lane = t & 31, ks = (t >> 5) % 18, ntile = t / (18 * 32);
    int h = lane >> 4, n = ntile * 16 + (lane & 15);
    int tap = ks >> 1, ky = tap / 3, kx = tap % 3;
    int cib = (ks & 1) << 5;
    _Float16* dst = Bp3 + (size_t)t * 16;
#pragma unroll
    for (int j = 0; j < 16; ++j) {
        int ci = cib + kmapB(j, h);
        dst[j] = (_Float16)w3[((n * 64 + ci) * 3 + ky) * 3 + kx];
    }
}

// ---------------------------------------------------------------------------
// conv2 implicit GEMM via WMMA: one wave = one 16x16 tile (M-tile = one output
// row of one image, since W2=16). A: 2x b128 loads/lane/K-step from NHWC-halo.
// D: transpose through LDS -> one b128 store per lane.
// ---------------------------------------------------------------------------
__global__ void conv2_wmma_kernel(const _Float16* __restrict__ x1, const _Float16* __restrict__ Bp2,
                                  const float* __restrict__ b2, _Float16* __restrict__ x2) {
    __shared__ alignas(16) _Float16 tile[8][256];
    int lane = threadIdx.x & 31;
    int wv = threadIdx.x >> 5;
    int wave = (blockIdx.x * blockDim.x + threadIdx.x) >> 5;   // 81920 waves exactly
    int ntile = wave & 3;
    int mtile = wave >> 2;
    int h = lane >> 4, lm = lane & 15;
    int m = mtile * 16 + lm;
    int img = m / (H2 * W2);
    int rem = m % (H2 * W2);
    int oy = rem / W2, ox = rem % W2;
    const _Float16* xin = x1 + (size_t)img * (H1P * W1P * 32);
    const _Float16* bfrag = Bp2 + ((size_t)(ntile * 9) * 32 + lane) * 16;
    v8f acc = {};
#pragma unroll
    for (int ks = 0; ks < 9; ++ks) {
        int ky = ks / 3, kx = ks % 3;
        // halo: (iy+1, ix+1) = (oy*2+ky, ox*2+kx)
        const _Float16* pix = xin + (((size_t)(oy * 2 + ky)) * W1P + (ox * 2 + kx)) * 32;
        v8h lo = *(const v8h*)(pix + 8 * h);
        v8h hi = *(const v8h*)(pix + 16 + 8 * h);
        v16h a = __builtin_shufflevector(lo, hi, 0, 1, 2, 3, 4, 5, 6, 7,
                                         8, 9, 10, 11, 12, 13, 14, 15);
        v16h b = *(const v16h*)(bfrag + (size_t)ks * 32 * 16);
        acc = __builtin_amdgcn_wmma_f32_16x16x32_f16(false, a, false, b, (short)0, acc, false, false);
    }
    // bias + relu + f16, transpose via LDS (channel-per-lane -> pixel-per-lane)
    int n = ntile * 16 + lm;
    float bias = b2[n];
#pragma unroll
    for (int r = 0; r < 8; ++r)
        tile[wv][(r + 8 * h) * 16 + lm] = (_Float16)fmaxf(acc[r] + bias, 0.0f);
    __syncthreads();
    int base_m = mtile * 16;
    int img0 = base_m / (H2 * W2);
    int oy0 = (base_m % (H2 * W2)) / W2;
    // lane -> pixel p=lm, channel-half h: 8 contiguous channels, one b128 store
    _Float16* dst = x2 + ((((size_t)img0 * H2P + (oy0 + 1)) * W2P + (lm + 1)) * 64)
                       + ntile * 16 + 8 * h;
    *(v8h*)dst = *(const v8h*)&tile[wv][lm * 16 + 8 * h];
}

// ---------------------------------------------------------------------------
// conv3 implicit GEMM via WMMA; writes f32 seq[img][c], c = n*40 + y*8 + x
// ---------------------------------------------------------------------------
__global__ void conv3_wmma_kernel(const _Float16* __restrict__ x2, const _Float16* __restrict__ Bp3,
                                  const float* __restrict__ b3, float* __restrict__ seq) {
    int lane = threadIdx.x & 31;
    int wave = (blockIdx.x * blockDim.x + threadIdx.x) >> 5;   // 10240 waves exactly
    int ntile = wave & 1;
    int mtile = wave >> 1;
    int h = lane >> 4, lm = lane & 15;
    int m = mtile * 16 + lm;
    int img = m / (H3 * W3);
    int rem = m % (H3 * W3);
    int oy = rem / W3, ox = rem % W3;
    const _Float16* xin = x2 + (size_t)img * (H2P * W2P * 64);
    const _Float16* bfrag = Bp3 + ((size_t)(ntile * 18) * 32 + lane) * 16;
    v8f acc = {};
    for (int ks = 0; ks < 18; ++ks) {
        int tap = ks >> 1;
        int ky = tap / 3, kx = tap % 3;
        int cib = (ks & 1) << 5;
        const _Float16* pix = xin + (((size_t)(oy * 2 + ky)) * W2P + (ox * 2 + kx)) * 64 + cib;
        v8h lo = *(const v8h*)(pix + 8 * h);
        v8h hi = *(const v8h*)(pix + 16 + 8 * h);
        v16h a = __builtin_shufflevector(lo, hi, 0, 1, 2, 3, 4, 5, 6, 7,
                                         8, 9, 10, 11, 12, 13, 14, 15);
        v16h b = *(const v16h*)(bfrag + (size_t)ks * 32 * 16);
        acc = __builtin_amdgcn_wmma_f32_16x16x32_f16(false, a, false, b, (short)0, acc, false, false);
    }
    int n = ntile * 16 + lm;
    float bias = b3[n];
#pragma unroll
    for (int r = 0; r < 8; ++r) {
        int row = mtile * 16 + r + 8 * h;
        int im = row / (H3 * W3), pos = row % (H3 * W3);
        seq[(size_t)im * SENSN + n * 40 + pos] = fmaxf(acc[r] + bias, 0.0f);
    }
}

// ---------------------------------------------------------------------------
// precompute softplus(w)*mask for sensory & recurrent synapse weights
// ---------------------------------------------------------------------------
__global__ void prep_ltc_kernel(const float* __restrict__ sw, const int* __restrict__ smask,
                                float* __restrict__ sw_pos,
                                const float* __restrict__ rw, const int* __restrict__ rmask,
                                float* __restrict__ rw_pos) {
    int t = blockIdx.x * blockDim.x + threadIdx.x;
    if (t < SENSN * UNITSN) sw_pos[t] = softplusf_(sw[t]) * (float)smask[t];
    if (t < UNITSN * UNITSN) rw_pos[t] = softplusf_(rw[t]) * (float)rmask[t];
}

// ---------------------------------------------------------------------------
// LTC cell: block per batch element; wave w owns units 6w..6w+5, lanes stride
// the 1283 sensory inputs; unfold loop on threads 0..47.
// ---------------------------------------------------------------------------
__global__ void ltc_kernel(const float* __restrict__ seq, const float* __restrict__ relpos,
                           const float* __restrict__ input_w, const float* __restrict__ input_b,
                           const float* __restrict__ sw_pos, const float* __restrict__ smu,
                           const float* __restrict__ ssig, const float* __restrict__ serev,
                           const float* __restrict__ rw_pos, const float* __restrict__ rmu,
                           const float* __restrict__ rsig, const float* __restrict__ rerev,
                           const float* __restrict__ gleak, const float* __restrict__ vleak,
                           const float* __restrict__ cm, const float* __restrict__ out_w,
                           const float* __restrict__ out_b, const float* __restrict__ head_w,
                           const float* __restrict__ head_b, float* __restrict__ out) {
    __shared__ float Is[SENSN];
    __shared__ float wnum[UNITSN], wden[UNITSN];
    __shared__ float v[UNITSN], vnew[UNITSN];
    __shared__ float pooled[MOTORN];
    int t = threadIdx.x;
    int b = blockIdx.x;
    int lane = t & 31, wv = t >> 5;
    if (t < UNITSN) v[t] = 0.0f;
    if (t < MOTORN) pooled[t] = 0.0f;
    __syncthreads();
    for (int s = 0; s < NS; ++s) {
        const float* xt = seq + (size_t)(b * NS + s) * SENSN;
        for (int i = t; i < SENSN; i += 256) {
            float xv = (i < 1280) ? xt[i] : relpos[(b * NS + s) * 3 + (i - 1280)];
            Is[i] = xv * input_w[i] + input_b[i];
        }
        __syncthreads();
        for (int uu = 0; uu < 6; ++uu) {
            int u = wv * 6 + uu;
            float an = 0.0f, ad = 0.0f;
            for (int i = lane; i < SENSN; i += 32) {
                int idx = i * UNITSN + u;
                float a = sw_pos[idx] * sigmoidf_(ssig[idx] * (Is[i] - smu[idx]));
                ad += a;
                an += a * serev[idx];
            }
            for (int off = 16; off > 0; off >>= 1) {
                an += __shfl_down(an, off, 32);
                ad += __shfl_down(ad, off, 32);
            }
            if (lane == 0) { wnum[u] = an; wden[u] = ad; }
        }
        __syncthreads();
        for (int f = 0; f < UNFOLDS; ++f) {
            if (t < UNITSN) {
                int u = t;
                float cmt = softplusf_(cm[u]) * (float)UNFOLDS;
                float gl = softplusf_(gleak[u]);
                float num = cmt * v[u] + gl * vleak[u] + wnum[u];
                float den = cmt + gl + wden[u];
                for (int u2 = 0; u2 < UNITSN; ++u2) {
                    int idx = u2 * UNITSN + u;
                    float a = rw_pos[idx] * sigmoidf_(rsig[idx] * (v[u2] - rmu[idx]));
                    num += a * rerev[idx];
                    den += a;
                }
                vnew[u] = num / (den + 1e-8f);
            }
            __syncthreads();
            if (t < UNITSN) v[t] = vnew[t];
            __syncthreads();
        }
        if (t < MOTORN) pooled[t] += v[t] * out_w[t] + out_b[t];
        __syncthreads();
    }
    if (t < NACT) {
        float acc = head_b[t];
        for (int mo = 0; mo < MOTORN; ++mo) acc += (pooled[mo] / (float)NS) * head_w[mo * NACT + t];
        out[b * NACT + t] = tanhf(acc);
    }
}

// ---------------------------------------------------------------------------

static inline size_t align256(size_t x) { return (x + 255) & ~(size_t)255; }

extern "C" void kernel_launch(void* const* d_in, const int* in_sizes, int n_in,
                              void* d_out, int out_size, void* d_ws, size_t ws_size,
                              hipStream_t stream) {
    const float* depth     = (const float*)d_in[0];
    const float* relpos    = (const float*)d_in[1];
    const float* w1        = (const float*)d_in[2];
    const float* b1        = (const float*)d_in[3];
    const float* w2        = (const float*)d_in[4];
    const float* b2        = (const float*)d_in[5];
    const float* w3        = (const float*)d_in[6];
    const float* b3        = (const float*)d_in[7];
    const float* input_w   = (const float*)d_in[8];
    const float* input_b   = (const float*)d_in[9];
    const float* sens_w    = (const float*)d_in[10];
    const float* sens_mu   = (const float*)d_in[11];
    const float* sens_sig  = (const float*)d_in[12];
    const float* sens_erev = (const float*)d_in[13];
    const float* rec_w     = (const float*)d_in[14];
    const float* rec_mu    = (const float*)d_in[15];
    const float* rec_sig   = (const float*)d_in[16];
    const float* rec_erev  = (const float*)d_in[17];
    const float* gleak     = (const float*)d_in[18];
    const float* vleak     = (const float*)d_in[19];
    const float* cm        = (const float*)d_in[20];
    const float* out_w     = (const float*)d_in[21];
    const float* out_b     = (const float*)d_in[22];
    const float* head_w    = (const float*)d_in[23];
    const float* head_b    = (const float*)d_in[24];
    const int*   smask     = (const int*)d_in[25];
    const int*   rmask     = (const int*)d_in[26];
    float* out = (float*)d_out;

    // workspace carve-up (256B aligned), ~166 MB (L2 is 192 MB)
    const size_t x1_halves = (size_t)NIMG * H1P * W1P * 32;   // 49,020,928
    const size_t x2_halves = (size_t)NIMG * H2P * W2P * 64;   // 28,311,552
    uint8_t* ws = (uint8_t*)d_ws;
    size_t off = 0;
    _Float16* x1 = (_Float16*)(ws + off); off += align256(x1_halves * 2);
    _Float16* x2 = (_Float16*)(ws + off); off += align256(x2_halves * 2);
    float* seq   = (float*)(ws + off);    off += align256((size_t)NB * NS * SENSN * 4);
    _Float16* Bp2 = (_Float16*)(ws + off); off += align256((size_t)4 * 9 * 32 * 16 * 2);
    _Float16* Bp3 = (_Float16*)(ws + off); off += align256((size_t)2 * 18 * 32 * 16 * 2);
    float* sw_pos = (float*)(ws + off);    off += align256((size_t)SENSN * UNITSN * 4);
    float* rw_pos = (float*)(ws + off);    off += align256((size_t)UNITSN * UNITSN * 4);
    (void)ws_size; (void)in_sizes; (void)n_in; (void)out_size;

    // zero the haloed activation buffers (16B per thread; sizes divide exactly)
    fill_zero_kernel<<<(unsigned)(x1_halves * 2 / 16 / 256), 256, 0, stream>>>((uint4*)x1);
    fill_zero_kernel<<<(unsigned)(x2_halves * 2 / 16 / 256), 256, 0, stream>>>((uint4*)x2);

    // weight pack + LTC param prep (tiny)
    pack_b2_kernel<<<5, 256, 0, stream>>>(w2, Bp2);
    pack_b3_kernel<<<5, 256, 0, stream>>>(w3, Bp3);
    prep_ltc_kernel<<<(SENSN * UNITSN + 255) / 256, 256, 0, stream>>>(
        sens_w, smask, sw_pos, rec_w, rmask, rw_pos);

    // conv pipeline
    conv1_kernel<<<(NIMG * H1 * W1) / 256, 256, 0, stream>>>(depth, w1, b1, x1);
    conv2_wmma_kernel<<<10240, 256, 0, stream>>>(x1, Bp2, b2, x2);   // 81920 waves
    conv3_wmma_kernel<<<1280, 256, 0, stream>>>(x2, Bp3, b3, seq);   // 10240 waves

    // recurrent LTC cell + head
    ltc_kernel<<<NB, 256, 0, stream>>>(seq, relpos, input_w, input_b,
                                       sw_pos, sens_mu, sens_sig, sens_erev,
                                       rw_pos, rec_mu, rec_sig, rec_erev,
                                       gleak, vleak, cm, out_w, out_b,
                                       head_w, head_b, out);
}
